// HouseholderTower_8461085573501
// MI455X (gfx1250) — matrix operand order repaired and checked by reading, not compiled, not run on hardware
//
#include <hip/hip_runtime.h>
#include <hip/hip_bf16.h>

typedef __attribute__((ext_vector_type(16))) __bf16          v16bf;
typedef __attribute__((ext_vector_type(16))) unsigned short  v16u;
typedef __attribute__((ext_vector_type(8)))  float           v8f;

static constexpr int kD   = 1024;   // feature dim (GEMM N and K)
static constexpr int kHHR = 32;     // number of Householder reflectors

__device__ __forceinline__ unsigned short f32_to_bf16_rne(float f) {
    unsigned u = __builtin_bit_cast(unsigned, f);
    u += 0x7FFFu + ((u >> 16) & 1u);   // round-to-nearest-even
    return (unsigned short)(u >> 16);
}

// ---------------------------------------------------------------------------
// Kernel 1: build P^T in bf16.  P = R0 R1 ... R31, Rk = I - v v^T / (v^T v).
// Row i of P is independent: p = e_i; p <- p - (p.v_k)/c_k * v_k  (k ascending).
// One 256-thread block per row; thread t owns elements 4t..4t+3.
// Output layout: PT[n * kD + r] = P[r][n]  (so GEMM B-lanes read contiguous K).
// ---------------------------------------------------------------------------
__global__ __launch_bounds__(256)
void build_pt_kernel(const float* __restrict__ V, unsigned short* __restrict__ PT) {
    __shared__ float s_hv[256];
    __shared__ float s_vv[256];
    const int t   = threadIdx.x;
    const int row = blockIdx.x;

    float p[4];
#pragma unroll
    for (int c = 0; c < 4; ++c) p[c] = (4 * t + c == row) ? 1.0f : 0.0f;

    for (int k = 0; k < kHHR; ++k) {
        const float4 v4 = *(const float4*)(V + (size_t)k * kD + 4 * t);
        const float vc[4] = {v4.x, v4.y, v4.z, v4.w};
        float hv = 0.f, vv = 0.f;
#pragma unroll
        for (int c = 0; c < 4; ++c) { hv += p[c] * vc[c]; vv += vc[c] * vc[c]; }
        s_hv[t] = hv; s_vv[t] = vv;
        __syncthreads();
        for (int off = 128; off > 0; off >>= 1) {
            if (t < off) { s_hv[t] += s_hv[t + off]; s_vv[t] += s_vv[t + off]; }
            __syncthreads();
        }
        const float scale = s_hv[0] / s_vv[0];
        __syncthreads();   // everyone has read s_hv[0]/s_vv[0] before next overwrite
#pragma unroll
        for (int c = 0; c < 4; ++c) p[c] -= scale * vc[c];
    }
#pragma unroll
    for (int c = 0; c < 4; ++c)
        PT[(size_t)(4 * t + c) * kD + row] = f32_to_bf16_rne(p[c]);
}

// ---------------------------------------------------------------------------
// Kernel 2: out = A @ P via v_wmma_f32_16x16x32_bf16, fp32 accumulation.
// Block = 256 threads = 8 waves. Wave w: rows [bx*128 + 16w, +16), cols
// [by*256, +256) -> 16 accumulator tiles (128 VGPRs/lane).
// A fragment (16x32 bf16): lanes 0-15 K 0..15, lanes 16-31 K
// 16..31; converted f32->bf16 on the fly.
// B fragment (32x16 bf16): lane n reads PT[n][k..k+15] contiguously
// (lanes 0-15 -> K 0..15, lanes 16-31 -> K 16..31).
// ---------------------------------------------------------------------------
__global__ __launch_bounds__(256)
void hht_gemm_kernel(const float* __restrict__ A,
                     const unsigned short* __restrict__ PT,
                     float* __restrict__ C) {
    const int lane = threadIdx.x & 31;
    const int wave = threadIdx.x >> 5;
    const int half = lane >> 4;          // 0: lanes 0-15, 1: lanes 16-31
    const int l15  = lane & 15;
    const int m0   = blockIdx.x * 128 + wave * 16;
    const int n0   = blockIdx.y * 256;

    v8f acc[16] = {};

    const float* arow = A + (size_t)(m0 + l15) * kD;

#pragma unroll 1
    for (int k = 0; k < kD; k += 32) {
        // ---- A fragment: two 32B chunks, K = kb+16*half .. kb+16*half+15 ----
        const float* ap = arow + k + half * 16;
        const v8f a_lo  = *(const v8f*)(ap);
        const v8f a_hi  = *(const v8f*)(ap + 8);
        // prefetch next A chunk (global_prefetch_b8)
        __builtin_prefetch(arow + k + 32, 0, 0);

        v16u au;
#pragma unroll
        for (int i = 0; i < 8; ++i) {
            au[i]     = f32_to_bf16_rne(a_lo[i]);
            au[i + 8] = f32_to_bf16_rne(a_hi[i]);
        }
        const v16bf afrag = __builtin_bit_cast(v16bf, au);

        // ---- 16 N-tiles reuse this A fragment ----
#pragma unroll
        for (int t = 0; t < 16; ++t) {
            const unsigned short* bp =
                PT + (size_t)(n0 + t * 16 + l15) * kD + k + half * 16;
            const v16u  bu    = *(const v16u*)(bp);       // 32B contiguous
            const v16bf bfrag = __builtin_bit_cast(v16bf, bu);
            acc[t] = __builtin_amdgcn_wmma_f32_16x16x32_bf16(
                false, afrag, false, bfrag, (short)0, acc[t], false, false);
        }
    }

    // ---- store: C VGPR r -> M = m0 + half*8 + r, N = n0 + t*16 + l15 ----
#pragma unroll
    for (int t = 0; t < 16; ++t) {
        const int col = n0 + t * 16 + l15;
#pragma unroll
        for (int r = 0; r < 8; ++r) {
            const int rowm = m0 + half * 8 + r;
            C[(size_t)rowm * kD + col] = acc[t][r];
        }
    }
}

// ---------------------------------------------------------------------------
extern "C" void kernel_launch(void* const* d_in, const int* in_sizes, int n_in,
                              void* d_out, int out_size, void* d_ws, size_t ws_size,
                              hipStream_t stream) {
    const float* node_feat = (const float*)d_in[0];   // [65536, 1024] f32
    const float* V         = (const float*)d_in[1];   // [32, 1024] f32
    float*       out       = (float*)d_out;           // [65536, 1024] f32
    unsigned short* PT     = (unsigned short*)d_ws;   // [1024, 1024] bf16 = 2 MB

    // 1) Compose the 32 reflectors into P^T (bf16), rows independent.
    build_pt_kernel<<<dim3(kD), dim3(256), 0, stream>>>(V, PT);

    // 2) out = node_feat @ P   (65536x1024x1024, bf16 WMMA, f32 accum)
    //    grid: M/128 x N/256 = 512 x 4
    hht_gemm_kernel<<<dim3(512, 4), dim3(256), 0, stream>>>(node_feat, PT, out);
}
